// ComplexWaveformSystem_38817914421256
// MI455X (gfx1250) — matrix operand order repaired and checked by reading, not compiled, not run on hardware
//
#include <hip/hip_runtime.h>

typedef __attribute__((ext_vector_type(16))) _Float16 v16h;
typedef __attribute__((ext_vector_type(8)))  float    v8f;

#define BATCH 8192
#define TOK   8192
#define LSEQ  128
#define LDA   136   // padded LDS row stride in halves: 272B = 68 dwords == 4 mod 64 banks

union FragU { v16h v; float4 q[2]; };

// ---------------------------------------------------------------------------
// Kernel 1: convert rx (f32) -> f16 workspace
// ---------------------------------------------------------------------------
__global__ void cw_cvt_kernel(const float* __restrict__ xr,
                              const float* __restrict__ xi,
                              _Float16* __restrict__ hr,
                              _Float16* __restrict__ hi, int n) {
    int idx = blockIdx.x * blockDim.x + threadIdx.x;
    if (idx < n) {
        hr[idx] = (_Float16)xr[idx];
        hi[idx] = (_Float16)xi[idx];
    }
}

// ---------------------------------------------------------------------------
// Kernel 2: bank = ifft_ortho(freq) via 128-point DFT with LDS twiddle table,
// emitted directly as f16. One block per token row.
//   bank[l] = (1/sqrt(L)) * sum_k (fr_k + i*fi_k) * e^{+2*pi*i*k*l/L}
// ---------------------------------------------------------------------------
__global__ __launch_bounds__(LSEQ) void cw_bank_kernel(
    const float* __restrict__ fr, const float* __restrict__ fi,
    _Float16* __restrict__ bkr, _Float16* __restrict__ bki) {
    __shared__ float sr[LSEQ], si[LSEQ], tc[LSEQ], ts[LSEQ];
    const int l = threadIdx.x;
    const int t = blockIdx.x;

    sr[l] = fr[t * LSEQ + l];
    si[l] = fi[t * LSEQ + l];
    float s, c;
    __sincosf((6.28318530717958647692f / (float)LSEQ) * (float)l, &s, &c);
    tc[l] = c;
    ts[l] = s;
    __syncthreads();

    float br = 0.f, bi = 0.f;
#pragma unroll 8
    for (int k = 0; k < LSEQ; ++k) {
        int idx = (k * l) & (LSEQ - 1);
        float cc = tc[idx], ss = ts[idx];
        br = fmaf(sr[k], cc, fmaf(-si[k], ss, br));
        bi = fmaf(sr[k], ss, fmaf(si[k], cc, bi));
    }
    const float scale = 0.08838834764831845f; // 1/sqrt(128), ortho norm
    bkr[t * LSEQ + l] = (_Float16)(br * scale);
    bki[t * LSEQ + l] = (_Float16)(bi * scale);
}

// ---------------------------------------------------------------------------
// Stage a [nrows x 128] f16 slab (row-major, ld=128) from global into LDS
// with padded row stride LDA. 16B-chunk granularity: global_load_b128 +
// ds_store_b128, coalesced across the 256-thread block.
// ---------------------------------------------------------------------------
__device__ __forceinline__ void stage_slab(const _Float16* __restrict__ g,
                                           int grow0, _Float16* __restrict__ s,
                                           int nrows, int tid) {
    const int nchunk = nrows * 16; // 16 x 16B chunks per 128-half row
    for (int idx = tid; idx < nchunk; idx += 256) {
        int r = idx >> 4;
        int c = idx & 15;
        *(float4*)(s + r * LDA + c * 8) =
            *(const float4*)(g + (size_t)(grow0 + r) * LSEQ + c * 8);
    }
}

// A-operand fragment (16x32 f16, contiguous K per lane half):
//   lane: M = lane&15, kh = lane>>4
//   VGPRs 0-3 <- K = k0 + kh*16 + 0..7       (16B load)
//   VGPRs 4-7 <- K = k0 + kh*16 + 8..15      (16B load)
__device__ __forceinline__ v16h load_fragA_lds(const _Float16* s, int row,
                                               int k0, int kh) {
    const _Float16* p = s + row * LDA + k0 + kh * 16;
    FragU u;
    u.q[0] = *(const float4*)(p);
    u.q[1] = *(const float4*)(p + 8);
    return u.v;
}

// B-operand fragment (32x16 f16, contiguous K per lane half; our B = bank^H so column N of
// B is row t of bank with K contiguous):
//   lane: N = lane&15, kh = lane>>4
//   VGPR v <- K = kh*16 + 2v, 2v+1  => halves 0..15 are ONE contiguous run
//   of 16 K-values starting at k0 + kh*16  (two adjacent 16B loads).
__device__ __forceinline__ v16h load_fragB_lds(const _Float16* s, int row,
                                               int k0, int kh) {
    const _Float16* p = s + row * LDA + k0 + kh * 16;
    FragU u;
    u.q[0] = *(const float4*)(p);
    u.q[1] = *(const float4*)(p + 8);
    return u.v;
}

// ---------------------------------------------------------------------------
// Kernel 3: complex-magnitude GEMM with v_wmma_f32_16x16x32_f16.
// Block = 256 threads = 8 waves arranged 4(rows) x 2(cols); block tile is
// 128 rows x 64 cols. Full K=128 slabs for A (rx) and B (bank) staged in LDS
// once (~102 KB of the 320 KB/WGP), then each wave computes a 32x32 patch
// (2x2 WMMA tiles) from LDS fragments.
//   real = sum rx_r*b_r + rx_i*b_i      (one accumulator, two WMMAs)
//   imag = sum rx_i*b_r - sum rx_r*b_i  (two accumulators, VALU subtract)
// ---------------------------------------------------------------------------
__global__ __launch_bounds__(256) void cw_gemm_kernel(
    const _Float16* __restrict__ rxr, const _Float16* __restrict__ rxi,
    const _Float16* __restrict__ bkr, const _Float16* __restrict__ bki,
    const float* __restrict__ temp, float* __restrict__ out) {
    // LDS: A slabs 128xLDA (r,i), B slabs 64xLDA (r,i) = 52224 halves (102 KB)
    __shared__ _Float16 smem[2 * 128 * LDA + 2 * 64 * LDA];
    _Float16* sAr = smem;
    _Float16* sAi = sAr + 128 * LDA;
    _Float16* sBr = sAi + 128 * LDA;
    _Float16* sBi = sBr + 64 * LDA;

    const int tid  = threadIdx.x;
    const int lane = tid & 31;
    const int wave = tid >> 5;
    const int wr   = wave >> 1;  // 0..3  (row wave)
    const int wc   = wave & 1;   // 0..1  (col wave)
    const int kh   = lane >> 4;
    const int ln   = lane & 15;

    const int blockRow0 = blockIdx.y * 128;  // batch rows
    const int blockCol0 = blockIdx.x * 64;   // token cols

    stage_slab(rxr, blockRow0, sAr, 128, tid);
    stage_slab(rxi, blockRow0, sAi, 128, tid);
    stage_slab(bkr, blockCol0, sBr, 64, tid);
    stage_slab(bki, blockCol0, sBi, 64, tid);
    __syncthreads();

    v8f accR[2][2] = {};  // rr + ii  (real)
    v8f accP[2][2] = {};  // rx_i * b_r
    v8f accQ[2][2] = {};  // rx_r * b_i

#pragma unroll
    for (int k0 = 0; k0 < LSEQ; k0 += 32) {
        v16h ar[2], ai[2], br[2], bi[2];
#pragma unroll
        for (int i = 0; i < 2; ++i) {
            ar[i] = load_fragA_lds(sAr, wr * 32 + i * 16 + ln, k0, kh);
            ai[i] = load_fragA_lds(sAi, wr * 32 + i * 16 + ln, k0, kh);
            br[i] = load_fragB_lds(sBr, wc * 32 + i * 16 + ln, k0, kh);
            bi[i] = load_fragB_lds(sBi, wc * 32 + i * 16 + ln, k0, kh);
        }
#pragma unroll
        for (int i = 0; i < 2; ++i) {
#pragma unroll
            for (int j = 0; j < 2; ++j) {
                accR[i][j] = __builtin_amdgcn_wmma_f32_16x16x32_f16(
                    false, ar[i], false, br[j], (short)0, accR[i][j], false, false);
                accR[i][j] = __builtin_amdgcn_wmma_f32_16x16x32_f16(
                    false, ai[i], false, bi[j], (short)0, accR[i][j], false, false);
                accP[i][j] = __builtin_amdgcn_wmma_f32_16x16x32_f16(
                    false, ai[i], false, br[j], (short)0, accP[i][j], false, false);
                accQ[i][j] = __builtin_amdgcn_wmma_f32_16x16x32_f16(
                    false, ar[i], false, bi[j], (short)0, accQ[i][j], false, false);
            }
        }
    }

    // v_rcp_f32 (<=1 ulp) instead of the full IEEE divide expansion; exact for
    // temperature==1 and far below the f16 quantization noise otherwise.
    const float invT = __builtin_amdgcn_rcpf(temp[0]);
#pragma unroll
    for (int i = 0; i < 2; ++i) {
#pragma unroll
        for (int j = 0; j < 2; ++j) {
            const int col     = blockCol0 + wc * 32 + j * 16 + ln;
            const int rowBase = blockRow0 + wr * 32 + i * 16 + kh * 8;
#pragma unroll
            for (int r = 0; r < 8; ++r) {
                float re = accR[i][j][r];
                float im = accP[i][j][r] - accQ[i][j][r];
                // Output is written once and never re-read: non-temporal.
                __builtin_nontemporal_store(
                    (re * re + im * im) * invT,
                    &out[(size_t)(rowBase + r) * TOK + col]);
            }
        }
    }
}

// ---------------------------------------------------------------------------
extern "C" void kernel_launch(void* const* d_in, const int* in_sizes, int n_in,
                              void* d_out, int out_size, void* d_ws, size_t ws_size,
                              hipStream_t stream) {
    const float* rx_r = (const float*)d_in[0];
    const float* rx_i = (const float*)d_in[1];
    const float* fq_r = (const float*)d_in[2];
    const float* fq_i = (const float*)d_in[3];
    const float* temp = (const float*)d_in[4];
    float* out = (float*)d_out;

    // Workspace: 4 f16 arrays of 8192*128 = 8 MB total
    _Float16* rxr = (_Float16*)d_ws;
    _Float16* rxi = rxr + (size_t)BATCH * LSEQ;
    _Float16* bkr = rxi + (size_t)BATCH * LSEQ;
    _Float16* bki = bkr + (size_t)TOK * LSEQ;

    const int n = BATCH * LSEQ;
    cw_cvt_kernel<<<(n + 255) / 256, 256, 0, stream>>>(rx_r, rx_i, rxr, rxi, n);
    cw_bank_kernel<<<TOK, LSEQ, 0, stream>>>(fq_r, fq_i, bkr, bki);
    cw_gemm_kernel<<<dim3(TOK / 64, BATCH / 128), 256, 0, stream>>>(
        rxr, rxi, bkr, bki, temp, out);
}
